// BaseAttention_40329742910040
// MI455X (gfx1250) — matrix operand order repaired and checked by reading, not compiled, not run on hardware
//
#include <hip/hip_runtime.h>
#include <cmath>

typedef __attribute__((ext_vector_type(16))) _Float16 v16h;
typedef __attribute__((ext_vector_type(8)))  _Float16 v8h;
typedef __attribute__((ext_vector_type(4)))  _Float16 v4h;
typedef __attribute__((ext_vector_type(8)))  float    v8f;
typedef __attribute__((ext_vector_type(4)))  float    v4f;

#define WMMA(A, Bm, C) \
    __builtin_amdgcn_wmma_f32_16x16x32_f16(false, (A), false, (Bm), (short)0, (C), false, false)

constexpr int Bb = 8, Hh = 16, Ss = 1024, Dd = 64;
constexpr int QT = 2;                       // 32 query rows per wave
constexpr int WAVES = 4;
constexpr float SCALE = 0.125f;             // 1/sqrt(64)
constexpr float MASKED_BIAS = -10000.0f;
constexpr size_t NELEM = (size_t)Bb * Hh * Ss * Dd;   // 8,388,608

// ---------------- Pass 1a: K f32 -> f16, same row-major layout ----------------
__global__ __launch_bounds__(256)
void cvt_k_kernel(const float* __restrict__ K, _Float16* __restrict__ Kh) {
    const size_t i4 = (size_t)blockIdx.x * 256 + threadIdx.x;   // 4 elems/thread
    const float4 x = ((const float4*)K)[i4];
    v4h y = { (_Float16)x.x, (_Float16)x.y, (_Float16)x.z, (_Float16)x.w };
    *(v4h*)(Kh + i4 * 4) = y;
}

// ------------- Pass 1b: V f32 -> f16 transposed: VT[bh][d][key] ---------------
__global__ __launch_bounds__(256)
void cvt_vt_kernel(const float* __restrict__ V, _Float16* __restrict__ VT) {
    __shared__ _Float16 tile[64 * 68];                  // 64x64, padded stride 68
    const int t  = threadIdx.x;
    const int bh = blockIdx.x >> 4;                     // 16 key-tiles per bh
    const int k0 = (blockIdx.x & 15) * 64;
    const float* vsrc = V + (size_t)bh * Ss * Dd + (size_t)k0 * Dd;
    _Float16*    vdst = VT + (size_t)bh * Dd * Ss + k0;
    #pragma unroll
    for (int i = 0; i < 16; ++i) {                      // coalesced reads (d fast)
        const int lin = i * 256 + t, key = lin >> 6, d = lin & 63;
        tile[key * 68 + d] = (_Float16)vsrc[(size_t)key * Dd + d];
    }
    __syncthreads();
    #pragma unroll
    for (int i = 0; i < 16; ++i) {                      // coalesced writes (key fast)
        const int lin = i * 256 + t, d = lin >> 6, key = lin & 63;
        vdst[(size_t)d * Ss + key] = tile[key * 68 + d];
    }
}

// --------------------- Main pass: flash attention, f16 K/V --------------------
__global__ __launch_bounds__(32 * WAVES)
void fa_fwd_f16_kernel(const float* __restrict__ Q,
                       const _Float16* __restrict__ Kh,
                       const _Float16* __restrict__ VT,
                       float* __restrict__ O)
{
    const int lane = threadIdx.x & 31;
    const int lm   = lane & 15;
    const int lh   = lane >> 4;

    // gw is wave-uniform by construction; pin it so bh/q0/jend and the loop /
    // mask branches become scalar (s_cbranch) instead of exec-masked.
    const int gw = __builtin_amdgcn_readfirstlane(blockIdx.x * WAVES + (threadIdx.x >> 5));
    const int tiles_per_bh = Ss / (16 * QT);            // 32
    const int bh = gw / tiles_per_bh;
    const int q0 = (gw % tiles_per_bh) * (16 * QT);

    const float*    qp = Q  + (size_t)bh * Ss * Dd;
    const _Float16* kp = Kh + (size_t)bh * Ss * Dd;
    const _Float16* vp = VT + (size_t)bh * Dd * Ss;
    float*          op = O  + (size_t)bh * Ss * Dd;

    // Q^T B-fragments (loop-invariant), converted once per wave
    v16h qf[QT][2];
    #pragma unroll
    for (int t = 0; t < QT; ++t) {
        const float* qr = qp + (size_t)(q0 + t * 16 + lm) * Dd + lh * 16;
        #pragma unroll
        for (int e = 0; e < 16; ++e) {
            qf[t][0][e] = (_Float16)qr[e];
            qf[t][1][e] = (_Float16)qr[32 + e];
        }
    }

    v8f oacc[QT][4];
    float mst[QT], lst[QT];
    #pragma unroll
    for (int t = 0; t < QT; ++t) {
        #pragma unroll
        for (int c = 0; c < 4; ++c) oacc[t][c] = (v8f){};
        mst[t] = -INFINITY;
        lst[t] = 0.0f;
    }

    const int jend = q0 + 16 * QT;                      // scalar causal bound
    for (int jb = 0; jb < jend; jb += 32) {
        // ---- K A-fragments: pure b128 f16 loads, no conversion
        v16h ka[2][2];
        #pragma unroll
        for (int jt = 0; jt < 2; ++jt) {
            const _Float16* kr = kp + (size_t)(jb + jt * 16 + lm) * Dd + lh * 16;
            #pragma unroll
            for (int kb = 0; kb < 2; ++kb) {
                v8h lo = *(const v8h*)(kr + kb * 32);
                v8h hi = *(const v8h*)(kr + kb * 32 + 8);
                ka[jt][kb] = __builtin_shufflevector(lo, hi,
                    0,1,2,3,4,5,6,7,8,9,10,11,12,13,14,15);
            }
        }

        // ---- issue all 8 S^T WMMAs back-to-back (independent XDL work)
        v8f st[QT][2];
        #pragma unroll
        for (int t = 0; t < QT; ++t) { st[t][0] = (v8f){}; st[t][1] = (v8f){}; }
        #pragma unroll
        for (int t = 0; t < QT; ++t) {
            st[t][0] = WMMA(ka[0][0], qf[t][0], st[t][0]);
            st[t][1] = WMMA(ka[1][0], qf[t][0], st[t][1]);
        }
        #pragma unroll
        for (int t = 0; t < QT; ++t) {
            st[t][0] = WMMA(ka[0][1], qf[t][1], st[t][0]);
            st[t][1] = WMMA(ka[1][1], qf[t][1], st[t][1]);
        }

        // ---- softmax for both q-tiles (fills WMMA hazard slots with real work)
        v16h bp[QT];
        #pragma unroll
        for (int t = 0; t < QT; ++t) {
            const int qi = q0 + t * 16 + lm;
            float p0[8], p1[8];
            float mx = -INFINITY;
            if (jb + 31 > q0 + t * 16) {                // scalar branch: diagonal only
                #pragma unroll
                for (int r = 0; r < 8; ++r) {
                    const int k0i = jb + 8 * lh + r;
                    const float x0 = (k0i      <= qi) ? st[t][0][r] * SCALE : MASKED_BIAS;
                    const float x1 = (k0i + 16 <= qi) ? st[t][1][r] * SCALE : MASKED_BIAS;
                    p0[r] = x0; p1[r] = x1;
                    mx = fmaxf(mx, fmaxf(x0, x1));
                }
            } else {                                    // interior: no mask VALU
                #pragma unroll
                for (int r = 0; r < 8; ++r) {
                    p0[r] = st[t][0][r] * SCALE;
                    p1[r] = st[t][1][r] * SCALE;
                    mx = fmaxf(mx, fmaxf(p0[r], p1[r]));
                }
            }
            mx = fmaxf(mx, __shfl_xor(mx, 16, 32));
            const float mn  = fmaxf(mst[t], mx);
            const float rsc = __expf(mst[t] - mn);      // first iter: exp(-inf)=0
            mst[t] = mn;

            float ls = 0.0f;
            #pragma unroll
            for (int r = 0; r < 8; ++r) {
                p0[r] = __expf(p0[r] - mn);
                p1[r] = __expf(p1[r] - mn);
                ls += p0[r] + p1[r];
            }
            lst[t] = lst[t] * rsc + ls + __shfl_xor(ls, 16, 32);

            #pragma unroll
            for (int c = 0; c < 4; ++c)
                #pragma unroll
                for (int r = 0; r < 8; ++r) oacc[t][c][r] *= rsc;

            // P^T C-layout -> B-layout: one half-wave swap
            #pragma unroll
            for (int r = 0; r < 8; ++r) {
                const float ownv  = lh ? p1[r] : p0[r];
                const float sendv = lh ? p0[r] : p1[r];
                const float othv  = __shfl_xor(sendv, 16, 32);
                bp[t][r]     = (_Float16)(lh ? othv : ownv);
                bp[t][8 + r] = (_Float16)(lh ? ownv : othv);
            }
        }

        // ---- V^T A-fragments (contiguous b128 loads) + 8 PV WMMAs
        #pragma unroll
        for (int c = 0; c < 4; ++c) {
            const _Float16* vr = vp + (size_t)(c * 16 + lm) * Ss + jb + lh * 16;
            v8h lo = *(const v8h*)(vr);
            v8h hi = *(const v8h*)(vr + 8);
            v16h va = __builtin_shufflevector(lo, hi,
                0,1,2,3,4,5,6,7,8,9,10,11,12,13,14,15);
            #pragma unroll
            for (int t = 0; t < QT; ++t)
                oacc[t][c] = WMMA(va, bp[t], oacc[t][c]);
        }

        if (jb + 32 < jend) {
            __builtin_prefetch(kp + (size_t)(jb + 32 + lm) * Dd, 0, 1);
            __builtin_prefetch(vp + (size_t)lm * Ss + jb + 32, 0, 1);
        }
    }

    #pragma unroll
    for (int t = 0; t < QT; ++t) {
        const float inv = 1.0f / lst[t];
        float* orow = op + (size_t)(q0 + t * 16 + lm) * Dd + 8 * lh;
        #pragma unroll
        for (int c = 0; c < 4; ++c) {
            v4f lo = { oacc[t][c][0] * inv, oacc[t][c][1] * inv,
                       oacc[t][c][2] * inv, oacc[t][c][3] * inv };
            v4f hi = { oacc[t][c][4] * inv, oacc[t][c][5] * inv,
                       oacc[t][c][6] * inv, oacc[t][c][7] * inv };
            *(v4f*)(orow + c * 16)     = lo;
            *(v4f*)(orow + c * 16 + 4) = hi;
        }
    }
}

// -------- Fallback: single-pass, f32 inputs converted in-kernel ---------------
__global__ __launch_bounds__(32 * WAVES)
void fa_fwd_f32_kernel(const float* __restrict__ Q,
                       const float* __restrict__ K,
                       const float* __restrict__ V,
                       float* __restrict__ O)
{
    const int lane = threadIdx.x & 31;
    const int lm   = lane & 15;
    const int lh   = lane >> 4;

    const int gw = __builtin_amdgcn_readfirstlane(blockIdx.x * WAVES + (threadIdx.x >> 5));
    const int tiles_per_bh = Ss / (16 * QT);
    const int bh = gw / tiles_per_bh;
    const int q0 = (gw % tiles_per_bh) * (16 * QT);

    const float* qp = Q + (size_t)bh * Ss * Dd;
    const float* kp = K + (size_t)bh * Ss * Dd;
    const float* vp = V + (size_t)bh * Ss * Dd;
    float*       op = O + (size_t)bh * Ss * Dd;

    v16h qf[QT][2];
    #pragma unroll
    for (int t = 0; t < QT; ++t) {
        const float* qr = qp + (size_t)(q0 + t * 16 + lm) * Dd + lh * 16;
        #pragma unroll
        for (int e = 0; e < 16; ++e) {
            qf[t][0][e] = (_Float16)qr[e];
            qf[t][1][e] = (_Float16)qr[32 + e];
        }
    }

    v8f oacc[QT][4];
    float mst[QT], lst[QT];
    #pragma unroll
    for (int t = 0; t < QT; ++t) {
        #pragma unroll
        for (int c = 0; c < 4; ++c) oacc[t][c] = (v8f){};
        mst[t] = -INFINITY;
        lst[t] = 0.0f;
    }

    const int jend = q0 + 16 * QT;
    for (int jb = 0; jb < jend; jb += 32) {
        v16h ka[2][2];
        #pragma unroll
        for (int jt = 0; jt < 2; ++jt) {
            const float* kr = kp + (size_t)(jb + jt * 16 + lm) * Dd;
            #pragma unroll
            for (int e = 0; e < 16; ++e) {
                const int d = lh * 16 + e;
                ka[jt][0][e] = (_Float16)kr[d];
                ka[jt][1][e] = (_Float16)kr[32 + d];
            }
        }

        v8f st[QT][2];
        #pragma unroll
        for (int t = 0; t < QT; ++t) { st[t][0] = (v8f){}; st[t][1] = (v8f){}; }
        #pragma unroll
        for (int t = 0; t < QT; ++t) {
            st[t][0] = WMMA(ka[0][0], qf[t][0], st[t][0]);
            st[t][1] = WMMA(ka[1][0], qf[t][0], st[t][1]);
        }
        #pragma unroll
        for (int t = 0; t < QT; ++t) {
            st[t][0] = WMMA(ka[0][1], qf[t][1], st[t][0]);
            st[t][1] = WMMA(ka[1][1], qf[t][1], st[t][1]);
        }

        v16h bp[QT];
        #pragma unroll
        for (int t = 0; t < QT; ++t) {
            const int qi = q0 + t * 16 + lm;
            float p0[8], p1[8];
            float mx = -INFINITY;
            #pragma unroll
            for (int r = 0; r < 8; ++r) {
                const int k0i = jb + 8 * lh + r;
                const float x0 = (k0i      <= qi) ? st[t][0][r] * SCALE : MASKED_BIAS;
                const float x1 = (k0i + 16 <= qi) ? st[t][1][r] * SCALE : MASKED_BIAS;
                p0[r] = x0; p1[r] = x1;
                mx = fmaxf(mx, fmaxf(x0, x1));
            }
            mx = fmaxf(mx, __shfl_xor(mx, 16, 32));
            const float mn  = fmaxf(mst[t], mx);
            const float rsc = __expf(mst[t] - mn);
            mst[t] = mn;

            float ls = 0.0f;
            #pragma unroll
            for (int r = 0; r < 8; ++r) {
                p0[r] = __expf(p0[r] - mn);
                p1[r] = __expf(p1[r] - mn);
                ls += p0[r] + p1[r];
            }
            lst[t] = lst[t] * rsc + ls + __shfl_xor(ls, 16, 32);

            #pragma unroll
            for (int c = 0; c < 4; ++c)
                #pragma unroll
                for (int r = 0; r < 8; ++r) oacc[t][c][r] *= rsc;

            #pragma unroll
            for (int r = 0; r < 8; ++r) {
                const float ownv  = lh ? p1[r] : p0[r];
                const float sendv = lh ? p0[r] : p1[r];
                const float othv  = __shfl_xor(sendv, 16, 32);
                bp[t][r]     = (_Float16)(lh ? othv : ownv);
                bp[t][8 + r] = (_Float16)(lh ? ownv : othv);
            }
        }

        #pragma unroll
        for (int c = 0; c < 4; ++c) {
            const float* vc = vp + (size_t)jb * Dd + c * 16 + lm;
            v16h va;
            #pragma unroll
            for (int e = 0; e < 16; ++e) {
                const int key = lh * 16 + e;
                va[e] = (_Float16)vc[(size_t)key * Dd];
            }
            #pragma unroll
            for (int t = 0; t < QT; ++t)
                oacc[t][c] = WMMA(va, bp[t], oacc[t][c]);
        }
    }

    #pragma unroll
    for (int t = 0; t < QT; ++t) {
        const float inv = 1.0f / lst[t];
        float* orow = op + (size_t)(q0 + t * 16 + lm) * Dd + 8 * lh;
        #pragma unroll
        for (int c = 0; c < 4; ++c) {
            v4f lo = { oacc[t][c][0] * inv, oacc[t][c][1] * inv,
                       oacc[t][c][2] * inv, oacc[t][c][3] * inv };
            v4f hi = { oacc[t][c][4] * inv, oacc[t][c][5] * inv,
                       oacc[t][c][6] * inv, oacc[t][c][7] * inv };
            *(v4f*)(orow + c * 16)     = lo;
            *(v4f*)(orow + c * 16 + 4) = hi;
        }
    }
}

extern "C" void kernel_launch(void* const* d_in, const int* in_sizes, int n_in,
                              void* d_out, int out_size, void* d_ws, size_t ws_size,
                              hipStream_t stream) {
    (void)in_sizes; (void)n_in; (void)out_size;
    const float* q = (const float*)d_in[0];
    const float* k = (const float*)d_in[1];
    const float* v = (const float*)d_in[2];
    float* out = (float*)d_out;

    const int tiles_per_bh = Ss / (16 * QT);                // 32
    const dim3 fa_grid((Bb * Hh * tiles_per_bh) / WAVES);   // 1024 blocks
    const dim3 fa_block(32 * WAVES);                        // 128 = 4 wave32

    const size_t need = 2 * NELEM * sizeof(_Float16);       // Kh + VT = 32 MiB
    if (ws_size >= need && d_ws != nullptr) {
        _Float16* kh = (_Float16*)d_ws;
        _Float16* vt = kh + NELEM;
        cvt_k_kernel<<<dim3(NELEM / 4 / 256), dim3(256), 0, stream>>>(k, kh);
        cvt_vt_kernel<<<dim3(Bb * Hh * (Ss / 64)), dim3(256), 0, stream>>>(v, vt);
        fa_fwd_f16_kernel<<<fa_grid, fa_block, 0, stream>>>(q, kh, vt, out);
    } else {
        fa_fwd_f32_kernel<<<fa_grid, fa_block, 0, stream>>>(q, k, v, out);
    }
}
